// GCN_45002667327799
// MI455X (gfx1250) — matrix operand order repaired and checked by reading, not compiled, not run on hardware
//
#include <hip/hip_runtime.h>
#include <hip/hip_bf16.h>
#include <math.h>

// Problem sizes (fixed by the reference)
#define BB 8
#define NN 4096
#define DD 256
#define VE_COLS 272        // 256 V cols + 1 denom col + 15 zero pad -> 17 tiles of 16
#define VE_PITCH 4096      // j pitch inside VeT (global)
#define CT 17              // column tiles of 16
#define LDS_PITCH 40       // f16 elements per col row in LDS (80B: 20-bank stride, conflict-minimal)
#define NCHUNK 1088        // 272 cols * 4 x 16B chunks per 32-j stage

typedef __attribute__((ext_vector_type(16))) _Float16 v16h;
typedef __attribute__((ext_vector_type(8)))  _Float16 v8h;
typedef __attribute__((ext_vector_type(8)))  float    v8f;
typedef int v4i __attribute__((ext_vector_type(4)));

#if __has_builtin(__builtin_amdgcn_global_load_async_to_lds_b128)
#define USE_ASYNC 1
#else
#define USE_ASYNC 0
#endif

__device__ __forceinline__ void async_copy16(const _Float16* src, _Float16* dst_lds) {
#if USE_ASYNC
  __builtin_amdgcn_global_load_async_to_lds_b128(
      (__attribute__((address_space(1))) v4i*)src,
      (__attribute__((address_space(3))) v4i*)dst_lds, 0, 0);
#else
  *(v8h*)dst_lds = *(const v8h*)src;   // global_load_b128 + ds_store_b128 fallback
#endif
}

__device__ __forceinline__ void wait_async_5() {
#if USE_ASYNC
#if __has_builtin(__builtin_amdgcn_s_wait_asynccnt)
  __builtin_amdgcn_s_wait_asynccnt(5);
#else
  asm volatile("s_wait_asynccnt 0x5" ::: "memory");
#endif
#endif
}

__device__ __forceinline__ void wait_async_0() {
#if USE_ASYNC
#if __has_builtin(__builtin_amdgcn_s_wait_asynccnt)
  __builtin_amdgcn_s_wait_asynccnt(0);
#else
  asm volatile("s_wait_asynccnt 0x0" ::: "memory");
#endif
#endif
}

// ---------------------------------------------------------------------------
// Kernel 1: k_score[b*N+j] = dot(K[b,j,:], W[0, D:2D]).  One wave per row.
// ---------------------------------------------------------------------------
__global__ __launch_bounds__(256) void gat_kscore(const float* __restrict__ K,
                                                  const float* __restrict__ W,
                                                  float* __restrict__ ks) {
  const int wave = threadIdx.x >> 5;
  const int lane = threadIdx.x & 31;
  const int row  = blockIdx.x * 8 + wave;            // grid = B*N/8
  const float4* kp = (const float4*)(K + (size_t)row * DD);
  const float4* wp = (const float4*)(W + DD);        // wk = second half of W
  float4 k0 = kp[lane], k1 = kp[lane + 32];
  float4 w0 = wp[lane], w1 = wp[lane + 32];
  float s = k0.x * w0.x + k0.y * w0.y + k0.z * w0.z + k0.w * w0.w
          + k1.x * w1.x + k1.y * w1.y + k1.z * w1.z + k1.w * w1.w;
#pragma unroll
  for (int off = 16; off > 0; off >>= 1) s += __shfl_xor(s, off, 32);
  if (lane == 0) ks[row] = s;
}

// ---------------------------------------------------------------------------
// Kernel 2: mx[b] = max_j ks[b,j]   (stability shift; cancels in the ratio)
// ---------------------------------------------------------------------------
__global__ __launch_bounds__(256) void gat_max(const float* __restrict__ ks,
                                               float* __restrict__ mx) {
  __shared__ float sm[256];
  const int b = blockIdx.x;
  float m = -3.0e38f;
  for (int i = threadIdx.x; i < NN; i += 256) m = fmaxf(m, ks[b * NN + i]);
  sm[threadIdx.x] = m;
  __syncthreads();
  for (int s = 128; s > 0; s >>= 1) {
    if (threadIdx.x < s) sm[threadIdx.x] = fmaxf(sm[threadIdx.x], sm[threadIdx.x + s]);
    __syncthreads();
  }
  if (threadIdx.x == 0) mx[b] = sm[0];
}

// ---------------------------------------------------------------------------
// Kernel 3: build VeT[b][col][j] (f16, transposed, padded):
//   col <  256 : e_j * V[b,j,col];  col == 256 : e_j;  col > 256 : 0
// where e_j = exp(ks[b,j] - mx[b]).
// ---------------------------------------------------------------------------
__global__ __launch_bounds__(256) void gat_build(const float* __restrict__ V,
                                                 const float* __restrict__ ks,
                                                 const float* __restrict__ mx,
                                                 _Float16* __restrict__ veT) {
  const int b  = blockIdx.x >> 4;                    // 16 j-blocks of 256 per batch
  const int j  = ((blockIdx.x & 15) << 8) + threadIdx.x;
  const float e = expf(ks[b * NN + j] - mx[b]);
  const float* vrow = V + ((size_t)b * NN + j) * DD;
  _Float16* vb = veT + (size_t)b * VE_COLS * VE_PITCH + j;
#pragma unroll 4
  for (int c = 0; c < DD; ++c) vb[(size_t)c * VE_PITCH] = (_Float16)(e * vrow[c]);
  vb[(size_t)DD * VE_PITCH] = (_Float16)e;
#pragma unroll
  for (int c = DD + 1; c < VE_COLS; ++c) vb[(size_t)c * VE_PITCH] = (_Float16)0.f;
}

// ---------------------------------------------------------------------------
// Kernel 4: out[b,i,:] = (adj[b,i,:] @ Ve) / (adj[b,i,:] @ e) via f16 WMMA.
// 8 waves/block; wave owns a 16-row i-tile + 17 resident v8f accumulators.
// B panels (32 j x 272 cols) staged once per block into LDS with
// GLOBAL_LOAD_ASYNC_TO_LDS_B128, triple-buffered, one barrier per K-step.
// ---------------------------------------------------------------------------
__global__ __launch_bounds__(256, 1) void gat_gemm(const float* __restrict__ adj,
                                                   const _Float16* __restrict__ veT,
                                                   float* __restrict__ out) {
  __shared__ _Float16 sB[3][VE_COLS * LDS_PITCH];    // 3 x 21.25 KB = 63.75 KB

  const int batch  = blockIdx.x >> 5;                // N/128 = 32 row-blocks/batch
  const int rowBlk = blockIdx.x & 31;
  const int tid    = threadIdx.x;
  const int wave   = tid >> 5;
  const int lane   = tid & 31;
  const int i0     = rowBlk * 128 + wave * 16;
  const int mrow   = lane & 15;
  const bool hi    = lane >= 16;

  const float*    aRow  = adj + ((size_t)batch * NN + i0 + mrow) * NN;
  const _Float16* vbase = veT + (size_t)batch * VE_COLS * VE_PITCH;

  // Issue the async copy of stage s (j0 = s*32) into sB[s%3].
  // Exactly 5 instructions per thread (uniform per wave -> ASYNCcnt += 5).
  auto issue = [&](int s) {
    const int j0 = s * 32;
    _Float16* buf = &sB[s % 3][0];
#pragma unroll
    for (int k = 0; k < 5; ++k) {
      int c = tid + k * 256;
      if (c >= NCHUNK) c = NCHUNK - 1;               // duplicate copy: same data, harmless
      const int col = c >> 2, jc = c & 3;            // 4 x 16B chunks per col
      const _Float16* src = vbase + (size_t)col * VE_PITCH + j0 + jc * 8;
      _Float16* dst = buf + col * LDS_PITCH + jc * 8;
      async_copy16(src, dst);
    }
  };

  v8f acc[CT];
#pragma unroll
  for (int c = 0; c < CT; ++c) acc[c] = (v8f){0.f, 0.f, 0.f, 0.f, 0.f, 0.f, 0.f, 0.f};

  // Per-stage compute: A fragment from adj (f32 -> f16), B fragments from LDS.
  auto compute = [&](int s) {
    const int j0 = s * 32;
    __builtin_prefetch(aRow + j0 + 64, 0, 0);        // next adj chunk -> global_prefetch_b8

    // ISA 16-bit A 16x32 layout: lanes 0-15 K={0..7,16..23}; lanes 16-31 K={8..15,24..31}
    const int o1 = j0 + (hi ? 8 : 0);
    float4 a0 = *(const float4*)(aRow + o1);
    float4 a1 = *(const float4*)(aRow + o1 + 4);
    float4 a2 = *(const float4*)(aRow + o1 + 16);
    float4 a3 = *(const float4*)(aRow + o1 + 20);
    v16h a;
    a[0]  = (_Float16)a0.x; a[1]  = (_Float16)a0.y; a[2]  = (_Float16)a0.z; a[3]  = (_Float16)a0.w;
    a[4]  = (_Float16)a1.x; a[5]  = (_Float16)a1.y; a[6]  = (_Float16)a1.z; a[7]  = (_Float16)a1.w;
    a[8]  = (_Float16)a2.x; a[9]  = (_Float16)a2.y; a[10] = (_Float16)a2.z; a[11] = (_Float16)a2.w;
    a[12] = (_Float16)a3.x; a[13] = (_Float16)a3.y; a[14] = (_Float16)a3.z; a[15] = (_Float16)a3.w;

    const _Float16* buf = &sB[s % 3][0];
    const int jloc = hi ? 8 : 0;                     // lane n<16: j-local {0..7,16..23}; n>=16: {8..15,24..31}
#pragma unroll
    for (int ct = 0; ct < CT; ++ct) {
      const int col = ct * 16 + mrow;
      const _Float16* bp = buf + col * LDS_PITCH + jloc;
      v8h b0 = *(const v8h*)(bp);                    // ds_load_b128
      v8h b1 = *(const v8h*)(bp + 16);
      v16h bb;
#pragma unroll
      for (int e = 0; e < 8; ++e) { bb[e] = b0[e]; bb[e + 8] = b1[e]; }
      acc[ct] = __builtin_amdgcn_wmma_f32_16x16x32_f16(
          false, a, false, bb, (short)0, acc[ct], false, false);
    }
  };

  issue(0);
  for (int s = 0; s < 127; ++s) {
    issue(s + 1);                                    // writes sB[(s+1)%3]
    wait_async_5();                                  // in-order: stage s resident
    __syncthreads();
    compute(s);                                      // laggards read s-1: (s+1)%3 != (s-1)%3
  }
  wait_async_0();
  __syncthreads();
  compute(127);

  // Epilogue: divide by denominator column (tile 16, N=0 holds col 256).
  const int srcLane = lane & 16;
#pragma unroll
  for (int r = 0; r < 8; ++r) {
    float den = __shfl(acc[16][r], srcLane, 32);
    float inv = (den > 0.f) ? (1.0f / den) : 0.f;
    const int rowM = i0 + (hi ? 8 + r : r);
    float* op = out + ((size_t)batch * NN + rowM) * DD + mrow;
#pragma unroll
    for (int ct = 0; ct < 16; ++ct) op[ct * 16] = acc[ct][r] * inv;
  }
}

// ---------------------------------------------------------------------------
// Launcher.  Inputs: 0=Q (cancels in softmax), 1=K, 2=V, 3=adj, 4=W, 5=b
// (cancels).  Workspace: ks | mx | VeT  (~18.1 MB total).
// ---------------------------------------------------------------------------
extern "C" void kernel_launch(void* const* d_in, const int* in_sizes, int n_in,
                              void* d_out, int out_size, void* d_ws, size_t ws_size,
                              hipStream_t stream) {
  const float* K   = (const float*)d_in[1];
  const float* V   = (const float*)d_in[2];
  const float* adj = (const float*)d_in[3];
  const float* W   = (const float*)d_in[4];
  float* out = (float*)d_out;

  char* ws = (char*)d_ws;
  float*    ks  = (float*)(ws);                      // B*N floats = 128 KB
  float*    mx  = (float*)(ws + 131072);             // B floats
  _Float16* veT = (_Float16*)(ws + 262144);          // B*272*4096 f16 = 17.8 MB

  gat_kscore<<<BB * NN / 8, 256, 0, stream>>>(K, W, ks);
  gat_max<<<BB, 256, 0, stream>>>(ks, mx);
  gat_build<<<BB * 16, 256, 0, stream>>>(V, ks, mx, veT);
  gat_gemm<<<BB * 32, 256, 0, stream>>>(adj, veT, out);
}
